// metalearning_conv_34102040330492
// MI455X (gfx1250) — matrix operand order repaired and checked by reading, compiled, not run on hardware
//
#include <hip/hip_runtime.h>
#include <hip/hip_bf16.h>

typedef __attribute__((ext_vector_type(16))) _Float16 v16h;
typedef __attribute__((ext_vector_type(8)))  _Float16 v8h;
typedef __attribute__((ext_vector_type(8)))  float    v8f;

#define KH 3
#define KW 3
#define CIN 32
#define COUT 64
#define KPROD (KH*KW*CIN*COUT)   // 18432
#define KB_COLS (KPROD + COUT)   // 18496
#define B_ 16
#define H_ 192
#define W_ 192

// ---------------------------------------------------------------------------
// Stage 1: hypernetwork. kb[b][j] = (((x@W1+b1)@W2+b2)@W3+b3)[b][j]
// Kernels written as f16 in [b][tap][cout][cin] (WMMA-A friendly), bias as f32.
// ---------------------------------------------------------------------------
__global__ __launch_bounds__(256)
void hyper_gen_kernel(const float* __restrict__ dense,  // (16,16)
                      const float* __restrict__ W1,     // (16,8)
                      const float* __restrict__ b1,     // (8)
                      const float* __restrict__ W2,     // (8,16)
                      const float* __restrict__ b2,     // (16)
                      const float* __restrict__ W3,     // (16,18496)
                      const float* __restrict__ b3,     // (18496)
                      _Float16* __restrict__ kernF16,   // (16,9,64,32)
                      float* __restrict__ bias)         // (16,64)
{
    const int b = blockIdx.y;

    // Redundantly compute h2[16] per thread (256 FMAs, trivial).
    float h1[8];
#pragma unroll
    for (int j = 0; j < 8; ++j) {
        float s = b1[j];
#pragma unroll
        for (int k = 0; k < 16; ++k) s += dense[b*16 + k] * W1[k*8 + j];
        h1[j] = s;
    }
    float h2[16];
#pragma unroll
    for (int j = 0; j < 16; ++j) {
        float s = b2[j];
#pragma unroll
        for (int k = 0; k < 8; ++k) s += h1[k] * W2[k*16 + j];
        h2[j] = s;
    }

    const int j = blockIdx.x * blockDim.x + threadIdx.x;
    if (j >= KB_COLS) return;

    float s = b3[j];
#pragma unroll
    for (int k = 0; k < 16; ++k) s += h2[k] * W3[k*KB_COLS + j];

    if (j < KPROD) {
        // j = ((tap)*32 + ci)*64 + co   (tap = kh*3+kw, reference HWIO layout)
        const int co  = j & 63;
        const int ci  = (j >> 6) & 31;
        const int tap = j >> 11;
        kernF16[((b*9 + tap)*64 + co)*32 + ci] = (_Float16)s;
    } else {
        bias[b*64 + (j - KPROD)] = s;
    }
}

// ---------------------------------------------------------------------------
// Stage 2: per-sample 3x3 conv as implicit GEMM with v_wmma_f32_16x16x32_f16.
// Grid: (y=192, b=16); block 128 = 4 waves; wave w owns x in [48w, 48w+48).
// LDS: input rows y-1..y+1 as f16, layout [dy][px(0..195)][cin], zero-padded.
// ---------------------------------------------------------------------------
__global__ __launch_bounds__(128)
void conv_wmma_kernel(const float* __restrict__ xin,       // (16,32,192,192)
                      const _Float16* __restrict__ kernF16,// (16,9,64,32)
                      const float* __restrict__ bias,      // (16,64)
                      float* __restrict__ out)             // (16,64,192,192)
{
    __shared__ _Float16 smem[3 * 196 * CIN];   // 37,632 bytes

    const int y   = blockIdx.x;
    const int b   = blockIdx.y;
    const int tid = threadIdx.x;

    // --- Stage input halo rows into LDS (f16, [dy][px][ci]) -----------------
    for (int lin = tid; lin < 3 * CIN * 196; lin += 128) {
        const int px = lin % 196;              // padded x index, gx = px-1
        const int ci = (lin / 196) % CIN;
        const int dy = lin / (196 * CIN);      // 0..2 -> gy = y+dy-1
        const int gy = y + dy - 1;
        const int gx = px - 1;
        float v = 0.0f;
        if ((unsigned)gy < (unsigned)H_ && (unsigned)gx < (unsigned)W_)
            v = xin[((b*CIN + ci)*H_ + gy)*W_ + gx];
        smem[(dy*196 + px)*CIN + ci] = (_Float16)v;
    }
    __syncthreads();

    const int lane  = tid & 31;
    const int wave  = tid >> 5;
    const int n     = lane & 15;   // column within 16-wide tile
    const int khalf = lane >> 4;   // K half-select per ISA fragment layout

    const v8f vzero = {0.f, 0.f, 0.f, 0.f, 0.f, 0.f, 0.f, 0.f};
    v8f acc[3][4];
#pragma unroll
    for (int xt = 0; xt < 3; ++xt)
#pragma unroll
        for (int ct = 0; ct < 4; ++ct) acc[xt][ct] = vzero;

    const int x0base = wave * 48;

    for (int tap = 0; tap < 9; ++tap) {
        const int kh = tap / 3;
        const int kw = tap % 3;

        // B fragments from LDS: lane holds n=lane%16, cin chunks at 8*khalf and 16+8*khalf.
        v16h Bfr[3];
#pragma unroll
        for (int xt = 0; xt < 3; ++xt) {
            const int px = x0base + 16*xt + n + kw;      // <= 193 < 196
            const _Float16* p = &smem[(kh*196 + px)*CIN + khalf*8];
            union { v16h v; v8h h[2]; } u;
            u.h[0] = *(const v8h*)(p);
            u.h[1] = *(const v8h*)(p + 16);
            Bfr[xt] = u.v;
        }

#pragma unroll
        for (int ct = 0; ct < 4; ++ct) {
            // A fragment from global (L2-resident, 36KB/sample):
            // cin chunks at 8*khalf and 16+8*khalf.
            const int cout = ct*16 + n;
            const _Float16* ap = &kernF16[((b*9 + tap)*64 + cout)*32 + khalf*8];
            union { v16h v; v8h h[2]; } ua;
            ua.h[0] = *(const v8h*)(ap);
            ua.h[1] = *(const v8h*)(ap + 16);
#pragma unroll
            for (int xt = 0; xt < 3; ++xt) {
                acc[xt][ct] = __builtin_amdgcn_wmma_f32_16x16x32_f16(
                    /*neg_a=*/false, ua.v, /*neg_b=*/false, Bfr[xt],
                    /*c_mod=*/(short)0, acc[xt][ct],
                    /*reuse_a=*/false, /*reuse_b=*/false);
            }
        }
    }

    // --- Writeback with bias (C/D layout: rows r+8*khalf, col n) ------------
#pragma unroll
    for (int ct = 0; ct < 4; ++ct) {
#pragma unroll
        for (int r = 0; r < 8; ++r) {
            const int cout = ct*16 + r + 8*khalf;
            const float bv = bias[b*64 + cout];
#pragma unroll
            for (int xt = 0; xt < 3; ++xt) {
                const int xo = x0base + 16*xt + n;
                out[((b*COUT + cout)*H_ + y)*W_ + xo] = acc[xt][ct][r] + bv;
            }
        }
    }
}

// ---------------------------------------------------------------------------
extern "C" void kernel_launch(void* const* d_in, const int* in_sizes, int n_in,
                              void* d_out, int out_size, void* d_ws, size_t ws_size,
                              hipStream_t stream) {
    const float* conv_input = (const float*)d_in[0];
    const float* dense      = (const float*)d_in[1];
    const float* W1         = (const float*)d_in[2];
    const float* b1         = (const float*)d_in[3];
    const float* W2         = (const float*)d_in[4];
    const float* b2         = (const float*)d_in[5];
    const float* W3         = (const float*)d_in[6];
    const float* b3         = (const float*)d_in[7];

    _Float16* kernF16 = (_Float16*)d_ws;                           // 589,824 B
    float*    bias    = (float*)((char*)d_ws + (size_t)B_*9*64*32*2);

    // Stage 1: generate per-sample kernels/biases (tiny).
    dim3 g1((KB_COLS + 255) / 256, B_);
    hyper_gen_kernel<<<g1, 256, 0, stream>>>(dense, W1, b1, W2, b2, W3, b3,
                                             kernF16, bias);

    // Stage 2: WMMA implicit-GEMM conv, one block per (row, sample).
    dim3 g2(H_, B_);
    conv_wmma_kernel<<<g2, 128, 0, stream>>>(conv_input, kernF16, bias,
                                             (float*)d_out);
}
